// BareKANLayer_70334384439347
// MI455X (gfx1250) — matrix operand order repaired and checked by reading, not compiled, not run on hardware
//
#include <hip/hip_runtime.h>

// ---------------------------------------------------------------------------
// KAN layer (PCHIP spline features) on gfx1250 via dense-GEMM reformulation.
//
//   out[b,o] = sum_d hermite(x[b,d]; C[o,d,:], S[o,d,:]) + bias[o]
//            = Phi[b, :] . W[:, o]      (Phi: B x 8192, 4 nonzeros per d-block)
//
// Stage 1: build W = [C ; h*S] in f16, pre-swizzled into WMMA B-fragment order.
// Stage 2: split-K=2 tiled WMMA GEMM (256 blocks for machine fill); Phi tiles
//          scattered into a double-buffered LDS tile (1 barrier per K-step);
//          sched_group_barrier pipeline overlaps A-fragment LDS clauses with
//          WMMA execution.
// Stage 3: deterministic fixed-order combine: out = (p0 + p1) + bias.
// ---------------------------------------------------------------------------

typedef __attribute__((ext_vector_type(16))) _Float16 v16h;
typedef __attribute__((ext_vector_type(8)))  _Float16 v8h;
typedef __attribute__((ext_vector_type(8)))  float    v8f;

#define BN 8192
#define DN 64
#define KN 64
#define ON 64
#define KSPLIT 2
#define DPB (DN / KSPLIT)    // 32 d-values per block
static __device__ constexpr float XMIN = -2.0f;
static __device__ constexpr float XMAX =  2.0f;

// sched_group_barrier masks
#define SG_WMMA     0x008
#define SG_VMEM_RD  0x020
#define SG_DS_READ  0x100
#define SG_DS_WRITE 0x200

// B-fragment storage order (this code's contiguous k order, same as loadA):
// for K-block kb (32 K rows), lane L holds N = L%16, K = e + 16*(L>=16),
// elements e = 0..15 contiguous.  Flat index:
//   Wfrag[(((d*4 + nt)*4 + kb)*32 + lane)*16 + e]
__device__ __forceinline__ size_t wfrag_index(int d, int kk, int o) {
    int nt   = o >> 4;
    int nl   = o & 15;
    int kb   = kk >> 5;
    int kl   = kk & 31;
    int half = kl >> 4;            // K >= 16 within the 32-block
    int e    = kl & 15;
    int lane = nl + (half << 4);
    return ((size_t)(((d * 4 + nt) * 4 + kb) * 32 + lane) << 4) + e;
}

// ---------------------------------------------------------------------------
// Stage 1: PCHIP slopes (streaming, one thread per (o,d)) + f16 pack.
// ---------------------------------------------------------------------------
__global__ __launch_bounds__(256) void build_wfrag_kernel(
    const float* __restrict__ coeffs, _Float16* __restrict__ Wfrag) {
    int tid = blockIdx.x * blockDim.x + threadIdx.x;
    if (tid >= ON * DN) return;
    int o = tid >> 6;
    int d = tid & 63;
    const float* y = coeffs + ((size_t)o * DN + d) * KN;
    const float hh = (XMAX - XMIN) / (float)(KN - 1);

    float y0 = y[0];
    float y1 = y[1];
    float dp = (y1 - y0) / hh;                  // delta[0]
    float delta0 = dp, delta1 = 0.f, delta_pen = 0.f;
    Wfrag[wfrag_index(d, 0, o)] = (_Float16)y0;

    float yk = y1;
    for (int k = 1; k <= KN - 2; ++k) {
        float yn = y[k + 1];
        float dn = (yn - yk) / hh;              // delta[k]
        if (k == 1)      delta1    = dn;
        if (k == KN - 2) delta_pen = dp;        // delta[K-3]
        float s = (dp * dn > 0.f) ? (2.f * dp * dn / (dp + dn + 1e-12f)) : 0.f;
        Wfrag[wfrag_index(d, k,      o)] = (_Float16)yk;
        Wfrag[wfrag_index(d, 64 + k, o)] = (_Float16)(hh * s);
        dp = dn;
        yk = yn;
    }
    float dlast = dp;                           // delta[K-2]
    float d0 = (3.f * delta0 - delta1) * 0.5f;
    d0 = (d0 * delta0 <= 0.f) ? 0.f : d0;
    if ((delta0 * delta1 < 0.f) && (fabsf(d0) > 3.f * fabsf(delta0)))
        d0 = 3.f * delta0;
    float dN = (3.f * dlast - delta_pen) * 0.5f;
    dN = (dN * dlast <= 0.f) ? 0.f : dN;
    if ((dlast * delta_pen < 0.f) && (fabsf(dN) > 3.f * fabsf(dlast)))
        dN = 3.f * dlast;

    Wfrag[wfrag_index(d, KN - 1,      o)] = (_Float16)yk;       // y[K-1]
    Wfrag[wfrag_index(d, 64,          o)] = (_Float16)(hh * d0);
    Wfrag[wfrag_index(d, 64 + KN - 1, o)] = (_Float16)(hh * dN);
}

// ---------------------------------------------------------------------------
// Stage 2: split-K GEMM.  Block = 128 threads (4 waves) = 64 samples x all 64
// outputs x 32 d-values.  Wave w owns N-tile nt=w and the 4 M-subtiles.
// ---------------------------------------------------------------------------
#define APAD 136            // 128 + 8 halfs: breaks bank conflicts, keeps 16B align

__global__ __launch_bounds__(128) void kan_wmma_kernel(
    const float* __restrict__ x, const _Float16* __restrict__ Wfrag,
    float* __restrict__ partial) {
    __shared__ alignas(16) _Float16 At[2][64][APAD];   // double-buffered Phi tile

    const int tid  = threadIdx.x;
    const int wave = tid >> 5;          // == N-tile index nt
    const int lane = tid & 31;
    const int r    = lane & 15;
    const int hi   = lane >> 4;
    const int mt   = blockIdx.x >> 1;   // M-tile index
    const int ks   = blockIdx.x & 1;    // K-split index
    const int d0   = ks * DPB;
    const int blockbase = mt * 64;
    float* pout = partial + (size_t)ks * BN * ON;

    // zero both Phi buffers
    for (int idx = tid; idx < 2 * 64 * APAD; idx += 128)
        (&At[0][0][0])[idx] = (_Float16)0.f;

    // scatter role: thread handles (sample sc_s, half sc_half)
    const int sc_s    = tid & 63;
    const int sc_half = tid >> 6;       // 0: value weights, 1: slope weights
    const float invh = (float)(KN - 1) / (XMAX - XMIN);
    const float* xg = x + (size_t)(blockbase + sc_s) * DN + d0;   // this thread's x row

    // per-dd Hermite / extrapolation weights for this thread's (sample, half)
    auto weights = [&](int dd, int& c0, float& w0, float& w1) {
        float xv = xg[dd];                      // global read; L0/L2 resident
        float t  = (xv - XMIN) * invh;
        int   i;
        if (xv < XMIN) {                        // linear extrapolation (left)
            i = 0;
            if (sc_half == 0) { w0 = 1.f; w1 = 0.f; }
            else              { w0 = t;   w1 = 0.f; }
        } else if (xv > XMAX) {                 // linear extrapolation (right)
            i = KN - 2;
            if (sc_half == 0) { w0 = 0.f; w1 = 1.f; }
            else              { w0 = 0.f; w1 = t - (float)(KN - 1); }
        } else {                                // Hermite basis
            i = (int)floorf(t);
            i = (i < 0) ? 0 : ((i > KN - 2) ? KN - 2 : i);
            float u  = t - (float)i;
            float u2 = u * u, u3 = u2 * u;
            if (sc_half == 0) { w0 =  2.f * u3 - 3.f * u2 + 1.f;
                                w1 = -2.f * u3 + 3.f * u2; }
            else              { w0 = u3 - 2.f * u2 + u;
                                w1 = u3 - u2; }
        }
        c0 = sc_half * 64 + i;                  // scatter columns c0, c0+1
    };

    __syncthreads();                            // zero-fill visible to all

    // prologue: scatter dd = 0 into buffer 0
    int c0_hist[2];
    {
        int c0; float w0, w1;
        weights(0, c0, w0, w1);
        At[0][sc_s][c0]     = (_Float16)w0;
        At[0][sc_s][c0 + 1] = (_Float16)w1;
        c0_hist[0] = c0;
        c0_hist[1] = 0;                         // buffer 1 is all-zero; benign
    }
    __syncthreads();

    v8f acc0 = {}, acc1 = {}, acc2 = {}, acc3 = {};

    // A-fragment loader (this code's contiguous k order, same as Wfrag B):
    // lane<16: e0..15 -> K 0..15
    // lane>=16: e0..15 -> K 16..31
    auto loadA = [&](const _Float16* abase, int m, int kb) -> v16h {
        const _Float16* rp = abase + (size_t)(m * 16 + r) * APAD + kb * 32 + 16 * hi;
        return __builtin_shufflevector(*(const v8h*)rp, *(const v8h*)(rp + 8),
               0, 1, 2, 3, 4, 5, 6, 7, 8, 9, 10, 11, 12, 13, 14, 15);
    };

    for (int dd = 0; dd < DPB; ++dd) {
        const int d   = d0 + dd;
        const int buf = dd & 1;

        // ---- B fragments for this d ----
        const _Float16* bbase =
            Wfrag + ((size_t)((d * 4 + wave) * 4) * 32) * 16;
        v16h b0 = *(const v16h*)(bbase + ((size_t)(0 * 32 + lane) << 4));
        v16h b1 = *(const v16h*)(bbase + ((size_t)(1 * 32 + lane) << 4));
        v16h b2 = *(const v16h*)(bbase + ((size_t)(2 * 32 + lane) << 4));
        v16h b3 = *(const v16h*)(bbase + ((size_t)(3 * 32 + lane) << 4));

        if (dd + 1 < DPB) {
            // prefetch next-d B region: 32 lanes x 128B = full 4KB wave tile
            const char* pref = (const char*)(Wfrag +
                ((size_t)(((d + 1) * 4 + wave) * 4) * 32) * 16) + lane * 128;
            __builtin_prefetch(pref, 0, 3);

            // un-scatter stale columns + scatter dd+1 into the other buffer
            const int nb = buf ^ 1;
            const int oc = c0_hist[nb];
            At[nb][sc_s][oc]     = (_Float16)0.f;
            At[nb][sc_s][oc + 1] = (_Float16)0.f;
            int c0; float w0, w1;
            weights(dd + 1, c0, w0, w1);
            At[nb][sc_s][c0]     = (_Float16)w0;
            At[nb][sc_s][c0 + 1] = (_Float16)w1;
            c0_hist[nb] = c0;
        }

        // ---- 16 WMMAs over 4 K-blocks ----
        const _Float16* abase = &At[buf][0][0];
#pragma unroll
        for (int kb = 0; kb < 4; ++kb) {
            v16h a0 = loadA(abase, 0, kb);
            v16h a1 = loadA(abase, 1, kb);
            v16h a2 = loadA(abase, 2, kb);
            v16h a3 = loadA(abase, 3, kb);
            v16h b = (kb == 0) ? b0 : (kb == 1) ? b1 : (kb == 2) ? b2 : b3;
            acc0 = __builtin_amdgcn_wmma_f32_16x16x32_f16(
                false, a0, false, b, (short)0, acc0, false, false);
            acc1 = __builtin_amdgcn_wmma_f32_16x16x32_f16(
                false, a1, false, b, (short)0, acc1, false, false);
            acc2 = __builtin_amdgcn_wmma_f32_16x16x32_f16(
                false, a2, false, b, (short)0, acc2, false, false);
            acc3 = __builtin_amdgcn_wmma_f32_16x16x32_f16(
                false, a3, false, b, (short)0, acc3, false, false);
        }

        // ---- scheduling pipeline for this iteration's region ----
        // B tiles first, scatter stores early, then DS-read clauses of the
        // NEXT K-block overlap the WMMA group of the current one.
        __builtin_amdgcn_sched_group_barrier(SG_VMEM_RD, 8, 0);  // B fragments
        __builtin_amdgcn_sched_group_barrier(SG_DS_WRITE, 4, 0); // scatter
        __builtin_amdgcn_sched_group_barrier(SG_DS_READ, 8, 0);  // kb0 A
        __builtin_amdgcn_sched_group_barrier(SG_DS_READ, 8, 0);  // kb1 A
        __builtin_amdgcn_sched_group_barrier(SG_WMMA,    4, 0);  // kb0 wmma
        __builtin_amdgcn_sched_group_barrier(SG_DS_READ, 8, 0);  // kb2 A
        __builtin_amdgcn_sched_group_barrier(SG_WMMA,    4, 0);  // kb1 wmma
        __builtin_amdgcn_sched_group_barrier(SG_DS_READ, 8, 0);  // kb3 A
        __builtin_amdgcn_sched_group_barrier(SG_WMMA,    4, 0);  // kb2 wmma
        __builtin_amdgcn_sched_group_barrier(SG_WMMA,    4, 0);  // kb3 wmma

        __syncthreads();    // readers done with At[buf]; dd+1 scatter visible
    }

    // ---- epilogue: C/D fragment is [vgpr v -> M = v + 8*hi, N = lane%16] ----
    const int o = wave * 16 + r;
#pragma unroll
    for (int m = 0; m < 4; ++m) {
        v8f acc = (m == 0) ? acc0 : (m == 1) ? acc1 : (m == 2) ? acc2 : acc3;
        int rowb = blockbase + m * 16 + hi * 8;
#pragma unroll
        for (int v = 0; v < 8; ++v)
            pout[(size_t)(rowb + v) * ON + o] = acc[v];
    }
}

// ---------------------------------------------------------------------------
// Stage 3: deterministic fixed-order combine: out = (p0 + p1) + bias.
// ---------------------------------------------------------------------------
__global__ __launch_bounds__(256) void combine_kernel(
    const float* __restrict__ partial, const float* __restrict__ bias,
    float* __restrict__ out) {
    int idx = blockIdx.x * blockDim.x + threadIdx.x;
    if (idx >= BN * ON) return;
    float p0 = partial[idx];
    float p1 = partial[(size_t)BN * ON + idx];
    out[idx] = (p0 + p1) + bias[idx & (ON - 1)];
}

// ---------------------------------------------------------------------------
extern "C" void kernel_launch(void* const* d_in, const int* in_sizes, int n_in,
                              void* d_out, int out_size, void* d_ws, size_t ws_size,
                              hipStream_t stream) {
    const float* x      = (const float*)d_in[0];   // (8192, 64)
    const float* coeffs = (const float*)d_in[1];   // (64, 64, 64)
    const float* bias   = (const float*)d_in[2];   // (64,)
    float*       out    = (float*)d_out;           // (8192, 64)
    _Float16*    Wfrag  = (_Float16*)d_ws;         // 1 MB at offset 0
    float* partial = (float*)((char*)d_ws + (size_t)DN * 128 * ON * sizeof(_Float16));
                                                   // 2 x 2 MB partials

    (void)in_sizes; (void)n_in; (void)out_size; (void)ws_size;

    build_wfrag_kernel<<<(ON * DN + 255) / 256, 256, 0, stream>>>(coeffs, Wfrag);
    kan_wmma_kernel<<<(BN / 64) * KSPLIT, 128, 0, stream>>>(x, Wfrag, partial);
    combine_kernel<<<(BN * ON + 255) / 256, 256, 0, stream>>>(partial, bias, out);
}
